// MixtureOfMambaModel_39101382262868
// MI455X (gfx1250) — matrix operand order repaired and checked by reading, not compiled, not run on hardware
//
#include <hip/hip_runtime.h>
#include <hip/hip_bf16.h>

#define DEV __device__ __forceinline__

typedef __attribute__((ext_vector_type(16))) __bf16 v16bf;
typedef __attribute__((ext_vector_type(8)))  __bf16 v8bf;
typedef __attribute__((ext_vector_type(8)))  float  v8f;

// ---- model dims ----
constexpr int B_ = 16, TV = 32, CV = 512, HW = 14, TA = 64, FA = 128, QL = 20;
constexpr int D = 1024, INNER = 2048, NS = 64, HID = 4096, E = 4, NL = 4, NC = 13;
constexpr int S = 1 + TV + TA + 1;   // 98
constexpr int R = B_ * S;            // 1568 rows

DEV unsigned bfbits(float f) {       // RNE f32 -> bf16 bit pattern (low 16)
  unsigned u = __builtin_bit_cast(unsigned, f);
  return (u + 0x7FFFu + ((u >> 16) & 1u)) >> 16;
}
DEV unsigned pk2(float lo, float hi) {   // pack two bf16: lo in [15:0], hi in [31:16]
  unsigned a = bfbits(lo);
  unsigned b = __builtin_bit_cast(unsigned, hi);
  b = (b + 0x7FFFu + ((b >> 16) & 1u)) & 0xFFFF0000u;
  return a | b;
}
DEV float sigm(float x) { return 1.0f / (1.0f + expf(-x)); }

// =====================================================================
// bf16 WMMA GEMM: C[M,N] (+)= rowScale? * ( act( A[M,K] @ B[K,N] + bias ) )
// block tile 32x64, 8 waves, each wave one 16x16 C tile
// K-step 64 => two v_wmma_f32_16x16x32_bf16 per barrier round
// =====================================================================
constexpr int TM = 32, TN = 64, TK = 64;
constexpr int LSTR = TK + 8;   // bf16 LDS stride: 144B rows (16B-aligned chunks)

template<bool ACCUM, bool GELU_EPI, bool ROWSCALE>
__global__ __launch_bounds__(256)
void gemm_wmma(const float* __restrict__ A, const float* __restrict__ Bw,
               const float* __restrict__ bias, float* __restrict__ C,
               const float* __restrict__ rowScale, int rsStride,
               int M, int N, int K) {
  __shared__ __bf16 As[TM * LSTR];
  __shared__ __bf16 Bs[TN * LSTR];        // column-major: Bs[n][k]
  const int tid  = threadIdx.x;
  const int lane = tid & 31;
  const int wave = tid >> 5;              // 0..7
  const int wm   = wave & 1;              // 2 tiles in M
  const int wn   = wave >> 1;             // 4 tiles in N
  const int l15  = lane & 15;
  const int half = lane >> 4;
  const int m0   = blockIdx.y * TM;
  const int n0   = blockIdx.x * TN;

  v8f acc = {};

  for (int k0 = 0; k0 < K; k0 += TK) {
    if (k0 + TK < K) {  // speculative prefetch of next tiles (global_prefetch_b8)
      int pr = m0 + (tid >> 3);
      if (pr < M) __builtin_prefetch(A + (size_t)pr * K + k0 + TK, 0, 3);
      __builtin_prefetch(Bw + (size_t)(k0 + TK + (tid >> 4)) * N + n0, 0, 3);
    }
    { // A tile 32x64: 8 floats/thread -> packed bf16, one ds_store_b128
      int r = tid >> 3, c = (tid & 7) * 8;
      int gr = m0 + r;
      float4 v0 = {0.f, 0.f, 0.f, 0.f}, v1 = {0.f, 0.f, 0.f, 0.f};
      if (gr < M) {
        const float* ap = A + (size_t)gr * K + k0 + c;
        v0 = *(const float4*)ap;
        v1 = *(const float4*)(ap + 4);
      }
      uint4 p;
      p.x = pk2(v0.x, v0.y); p.y = pk2(v0.z, v0.w);
      p.z = pk2(v1.x, v1.y); p.w = pk2(v1.z, v1.w);
      *(uint4*)&As[r * LSTR + c] = p;
    }
    // B tile 64x64 row-major -> Bs col-major; adjacent k pairs packed to b32
    for (int j = 0; j < 2; ++j) {
      int kp = (tid >> 4) + j * 16;        // k-pair 0..31 -> k = 2*kp, 2*kp+1
      int nn = (tid & 15) * 4;
      const float* bp = Bw + (size_t)(k0 + 2 * kp) * N + n0 + nn;
      float4 u0 = *(const float4*)bp;
      float4 u1 = *(const float4*)(bp + N);
      *(unsigned*)&Bs[(nn + 0) * LSTR + 2 * kp] = pk2(u0.x, u1.x);
      *(unsigned*)&Bs[(nn + 1) * LSTR + 2 * kp] = pk2(u0.y, u1.y);
      *(unsigned*)&Bs[(nn + 2) * LSTR + 2 * kp] = pk2(u0.z, u1.z);
      *(unsigned*)&Bs[(nn + 3) * LSTR + 2 * kp] = pk2(u0.w, u1.w);
    }
    __syncthreads();

#pragma unroll
    for (int kb = 0; kb < 2; ++kb) {
      // A fragment: lane<16 -> K0..7,K16..23 ; lane>=16 -> K8..15,K24..31
      const __bf16* arow = &As[(wm * 16 + l15) * LSTR + kb * 32];
      v8bf a_lo = *(const v8bf*)(arow + half * 8);
      v8bf a_hi = *(const v8bf*)(arow + 16 + half * 8);
      v16bf av = __builtin_shufflevector(a_lo, a_hi, 0,1,2,3,4,5,6,7,8,9,10,11,12,13,14,15);
      // B fragment: lane<16 -> K0..7,K16..23 of column ; lane>=16 -> K8..15,K24..31
      const __bf16* bcol = &Bs[(wn * 16 + l15) * LSTR + kb * 32];
      v8bf b_lo = *(const v8bf*)(bcol + half * 8);
      v8bf b_hi = *(const v8bf*)(bcol + 16 + half * 8);
      v16bf bv = __builtin_shufflevector(b_lo, b_hi, 0,1,2,3,4,5,6,7,8,9,10,11,12,13,14,15);

      acc = __builtin_amdgcn_wmma_f32_16x16x32_bf16(false, av, false, bv,
                                                    (short)0, acc, false, false);
    }
    __syncthreads();
  }

  const int col = n0 + wn * 16 + l15;
  const float bv = bias ? bias[col] : 0.0f;
  for (int r = 0; r < 8; ++r) {
    int row = m0 + wm * 16 + r + 8 * half;   // VGPR r: M=r (lanes<16), M=r+8 (lanes>=16)
    if (row < M) {
      float v = acc[r] + bv;
      if (GELU_EPI) v = 0.5f * v * (1.0f + erff(v * 0.70710678118654752440f));
      if (ROWSCALE) v *= rowScale[(size_t)row * rsStride];
      float* p = C + (size_t)row * N + col;
      if (ACCUM) *p += v; else *p = v;
    }
  }
}

// =====================================================================
// small VALU kernels
// =====================================================================
__global__ void video_pool_k(const float* __restrict__ video, float* __restrict__ vpool) {
  int gw = (blockIdx.x * blockDim.x + threadIdx.x) >> 5;   // one wave per (b,tv,cv)
  int lane = threadIdx.x & 31;
  if (gw >= B_ * TV * CV) return;
  const float* src = video + (size_t)gw * (HW * HW);
  float s = 0.f;
  for (int i = lane; i < HW * HW; i += 32) s += src[i];
  for (int o = 16; o > 0; o >>= 1) s += __shfl_down(s, o, 32);
  if (lane == 0) vpool[gw] = s * (1.0f / (HW * HW));
}

__global__ void qmean_k(const float* __restrict__ q_emb, const int* __restrict__ question,
                        float* __restrict__ qm) {
  int i = blockIdx.x * blockDim.x + threadIdx.x;
  if (i >= B_ * D) return;
  int b = i >> 10, d = i & (D - 1);
  float s = 0.f;
  for (int t = 0; t < QL; ++t) s += q_emb[(size_t)question[b * QL + t] * D + d];
  qm[i] = s * (1.0f / QL);
}

__global__ void assemble_x_k(const float* __restrict__ vproj, const float* __restrict__ aproj,
                             const float* __restrict__ qproj, const float* __restrict__ cls,
                             const float* __restrict__ mod, const float* __restrict__ pos,
                             float* __restrict__ x) {
  int r = blockIdx.x;                       // 0..R-1
  int b = r / S, s = r % S;
  for (int d = threadIdx.x; d < D; d += blockDim.x) {
    float v; int m;
    if (s == 0)            { v = cls[d];                                         m = 3; }
    else if (s <= TV)      { v = vproj[((size_t)b * TV + (s - 1)) * D + d];      m = 0; }
    else if (s <= TV + TA) { v = aproj[((size_t)b * TA + (s - 1 - TV)) * D + d]; m = 1; }
    else                   { v = qproj[(size_t)b * D + d];                       m = 2; }
    x[(size_t)r * D + d] = v + mod[m * D + d] + pos[s * D + d];
  }
}

__global__ __launch_bounds__(256)
void rmsnorm_k(const float* __restrict__ x, const float* __restrict__ w,
               float* __restrict__ out) {
  __shared__ float red[256];
  int r = blockIdx.x;
  const float* xr = x + (size_t)r * D;
  float s = 0.f;
  for (int d = threadIdx.x; d < D; d += 256) { float v = xr[d]; s += v * v; }
  red[threadIdx.x] = s; __syncthreads();
  for (int o = 128; o > 0; o >>= 1) {
    if (threadIdx.x < o) red[threadIdx.x] += red[threadIdx.x + o];
    __syncthreads();
  }
  float inv = rsqrtf(red[0] * (1.0f / D) + 1e-6f);
  for (int d = threadIdx.x; d < D; d += 256)
    out[(size_t)r * D + d] = xr[d] * inv * w[d];
}

// depthwise causal conv (k=3, pad 2 left) + bias + silu; input = xz[:, :INNER]
__global__ void conv_silu_k(const float* __restrict__ xz, const float* __restrict__ cw,
                            const float* __restrict__ cb, float* __restrict__ xm) {
  int i = blockIdx.x * 256 + threadIdx.x;
  if (i >= R * INNER) return;
  int r = i / INNER, c = i - r * INNER;
  int s = r % S;
  const float* row = xz + (size_t)r * (2 * INNER) + c;
  float x0 = row[0];
  float x1 = (s >= 1) ? *(row - 2 * INNER) : 0.f;
  float x2 = (s >= 2) ? *(row - 4 * INNER) : 0.f;
  float v = cw[c * 3 + 0] * x2 + cw[c * 3 + 1] * x1 + cw[c * 3 + 2] * x0 + cb[c];
  xm[i] = v * sigm(v);
}

// serial linear recurrence over time per (b, n)
__global__ void ssm_scan_k(const float* __restrict__ dtp, const float* __restrict__ bp,
                           const float* __restrict__ cp, float* __restrict__ y) {
  int i = blockIdx.x * blockDim.x + threadIdx.x;
  if (i >= B_ * NS) return;
  int b = i / NS, n = i - b * NS;
  float st = 0.f;
  size_t base = (size_t)b * S * NS + n;
  for (int s = 0; s < S; ++s) {
    size_t idx = base + (size_t)s * NS;
    float d = sigm(dtp[idx]);
    st = (1.0f - d) * st + d * bp[idx];
    y[idx] = cp[idx] * st;
  }
}

// layernorm over NS=64 (no affine); one wave per row, 2 elems/lane
__global__ void ln64_k(const float* __restrict__ y, float* __restrict__ o) {
  int gw = (blockIdx.x * blockDim.x + threadIdx.x) >> 5;
  int lane = threadIdx.x & 31;
  if (gw >= R) return;
  const float* yr = y + (size_t)gw * NS;
  float a = yr[lane], b = yr[lane + 32];
  float s = a + b;
  for (int off = 16; off > 0; off >>= 1) s += __shfl_down(s, off, 32);
  s = __shfl(s, 0, 32);
  float mu = s * (1.0f / NS);
  float da = a - mu, db = b - mu;
  float vs = da * da + db * db;
  for (int off = 16; off > 0; off >>= 1) vs += __shfl_down(vs, off, 32);
  vs = __shfl(vs, 0, 32);
  float inv = rsqrtf(vs * (1.0f / NS) + 1e-5f);
  o[(size_t)gw * NS + lane] = da * inv;
  o[(size_t)gw * NS + lane + 32] = db * inv;
}

// y2 = (y2 + D*xm) * sigmoid(gate), gate = xz[:, INNER:]
__global__ void gate_mix_k(float* __restrict__ y2, const float* __restrict__ xm,
                           const float* __restrict__ xz, const float* __restrict__ Dp) {
  int i = blockIdx.x * 256 + threadIdx.x;
  if (i >= R * INNER) return;
  int r = i / INNER, c = i - r * INNER;
  float g = xz[(size_t)r * (2 * INNER) + INNER + c];
  y2[i] = (y2[i] + Dp[c] * xm[i]) * sigm(g);
}

// top-2-of-4 routing -> dense weights w[R][E]; one wave per row
__global__ void routing_k(const float* __restrict__ xn, const float* __restrict__ gw,
                          const float* __restrict__ gb, float* __restrict__ w) {
  int row = (blockIdx.x * blockDim.x + threadIdx.x) >> 5;
  int lane = threadIdx.x & 31;
  if (row >= R) return;
  const float* xr = xn + (size_t)row * D;
  float a0 = 0, a1 = 0, a2 = 0, a3 = 0;
  for (int d = lane; d < D; d += 32) {
    float xv = xr[d];
    const float* g = gw + (size_t)d * E;
    a0 += xv * g[0]; a1 += xv * g[1]; a2 += xv * g[2]; a3 += xv * g[3];
  }
  for (int o = 16; o > 0; o >>= 1) {
    a0 += __shfl_down(a0, o, 32); a1 += __shfl_down(a1, o, 32);
    a2 += __shfl_down(a2, o, 32); a3 += __shfl_down(a3, o, 32);
  }
  if (lane == 0) {
    float lg[4] = {a0 + gb[0], a1 + gb[1], a2 + gb[2], a3 + gb[3]};
    int i0 = 0;
    for (int e = 1; e < 4; ++e) if (lg[e] > lg[i0]) i0 = e;
    int i1 = -1;
    for (int e = 0; e < 4; ++e) { if (e == i0) continue; if (i1 < 0 || lg[e] > lg[i1]) i1 = e; }
    float m = fmaxf(lg[i0], lg[i1]);
    float e0 = expf(lg[i0] - m), e1 = expf(lg[i1] - m);
    float inv = 1.0f / (e0 + e1);
    float* wr = w + (size_t)row * E;
    wr[0] = 0.f; wr[1] = 0.f; wr[2] = 0.f; wr[3] = 0.f;
    wr[i0] = e0 * inv; wr[i1] = e1 * inv;
  }
}

// final rmsnorm(x[:,0]) @ head_w + head_b  (13 cols, tiny)
__global__ __launch_bounds__(256)
void head_k(const float* __restrict__ x, const float* __restrict__ fw,
            const float* __restrict__ hw, const float* __restrict__ hb,
            float* __restrict__ out) {
  __shared__ float red[256];
  __shared__ float xs[D];
  int b = blockIdx.x;
  const float* xr = x + (size_t)b * S * D;      // s == 0 row
  float s = 0.f;
  for (int d = threadIdx.x; d < D; d += 256) { float v = xr[d]; xs[d] = v; s += v * v; }
  red[threadIdx.x] = s; __syncthreads();
  for (int o = 128; o > 0; o >>= 1) {
    if (threadIdx.x < o) red[threadIdx.x] += red[threadIdx.x + o];
    __syncthreads();
  }
  float inv = rsqrtf(red[0] * (1.0f / D) + 1e-6f);
  for (int c = threadIdx.x; c < NC; c += 256) {
    float acc = hb[c];
    for (int d = 0; d < D; ++d) acc += xs[d] * inv * fw[d] * hw[d * NC + c];
    out[b * NC + c] = acc;
  }
}

// =====================================================================
// host side
// =====================================================================
static void gemm_launch(const float* A, const float* Bw, const float* bias, float* C,
                        const float* rs, int rsStride, int M, int N, int K,
                        int mode, hipStream_t st) {
  dim3 grid(N / TN, (M + TM - 1) / TM), blk(256);
  switch (mode) {
    case 0: gemm_wmma<false, false, false><<<grid, blk, 0, st>>>(A, Bw, bias, C, rs, rsStride, M, N, K); break;
    case 1: gemm_wmma<true,  false, false><<<grid, blk, 0, st>>>(A, Bw, bias, C, rs, rsStride, M, N, K); break; // accumulate
    case 2: gemm_wmma<false, true,  false><<<grid, blk, 0, st>>>(A, Bw, bias, C, rs, rsStride, M, N, K); break; // gelu
    case 3: gemm_wmma<true,  false, true ><<<grid, blk, 0, st>>>(A, Bw, bias, C, rs, rsStride, M, N, K); break; // accum+rowscale
  }
}

// workspace layout (floats)
constexpr size_t OFF_VPOOL = 0;
constexpr size_t OFF_VPROJ = OFF_VPOOL + (size_t)B_ * TV * CV;
constexpr size_t OFF_APROJ = OFF_VPROJ + (size_t)B_ * TV * D;
constexpr size_t OFF_QMEAN = OFF_APROJ + (size_t)B_ * TA * D;
constexpr size_t OFF_QPROJ = OFF_QMEAN + (size_t)B_ * D;
constexpr size_t OFF_X     = OFF_QPROJ + (size_t)B_ * D;
constexpr size_t OFF_XN    = OFF_X  + (size_t)R * D;
constexpr size_t OFF_XZ    = OFF_XN + (size_t)R * D;          // also reused as MoE hidden h
constexpr size_t OFF_XM    = OFF_XZ + (size_t)R * 2 * INNER;
constexpr size_t OFF_DT    = OFF_XM + (size_t)R * INNER;
constexpr size_t OFF_BP    = OFF_DT + (size_t)R * NS;
constexpr size_t OFF_CP    = OFF_BP + (size_t)R * NS;
constexpr size_t OFF_Y     = OFF_CP + (size_t)R * NS;
constexpr size_t OFF_YLN   = OFF_Y  + (size_t)R * NS;
constexpr size_t OFF_Y2    = OFF_YLN + (size_t)R * NS;
constexpr size_t OFF_RW    = OFF_Y2 + (size_t)R * INNER;

extern "C" void kernel_launch(void* const* d_in, const int* in_sizes, int n_in,
                              void* d_out, int out_size, void* d_ws, size_t ws_size,
                              hipStream_t stream) {
  (void)in_sizes; (void)n_in; (void)out_size; (void)ws_size;
  const float* video   = (const float*)d_in[0];
  const float* audio   = (const float*)d_in[1];
  const int*   question= (const int*)  d_in[2];
  const float* vproj_w = (const float*)d_in[3];
  const float* vproj_b = (const float*)d_in[4];
  const float* aproj_w = (const float*)d_in[5];
  const float* aproj_b = (const float*)d_in[6];
  const float* q_emb   = (const float*)d_in[7];
  const float* qproj_w = (const float*)d_in[8];
  const float* qproj_b = (const float*)d_in[9];
  const float* cls_tok = (const float*)d_in[10];
  const float* mod_emb = (const float*)d_in[11];
  const float* pos_emb = (const float*)d_in[12];
  const float* norm1_w = (const float*)d_in[13];
  const float* norm2_w = (const float*)d_in[14];
  const float* in_w    = (const float*)d_in[15];
  const float* in_b    = (const float*)d_in[16];
  const float* conv_w  = (const float*)d_in[17];
  const float* conv_b  = (const float*)d_in[18];
  const float* dt_w    = (const float*)d_in[19];
  const float* dt_b    = (const float*)d_in[20];
  const float* Bp_w    = (const float*)d_in[21];
  const float* Bp_b    = (const float*)d_in[22];
  const float* Cp_w    = (const float*)d_in[23];
  const float* Cp_b    = (const float*)d_in[24];
  const float* s2i_w   = (const float*)d_in[25];
  const float* s2i_b   = (const float*)d_in[26];
  const float* D_par   = (const float*)d_in[27];
  const float* out_w   = (const float*)d_in[28];
  const float* out_b   = (const float*)d_in[29];
  const float* gate_w  = (const float*)d_in[30];
  const float* gate_b  = (const float*)d_in[31];
  const float* e_w1    = (const float*)d_in[32];
  const float* e_b1    = (const float*)d_in[33];
  const float* e_w2    = (const float*)d_in[34];
  const float* e_b2    = (const float*)d_in[35];
  const float* fnorm_w = (const float*)d_in[36];
  const float* head_w  = (const float*)d_in[37];
  const float* head_b  = (const float*)d_in[38];

  float* W = (float*)d_ws;
  float* vpool = W + OFF_VPOOL;
  float* vproj = W + OFF_VPROJ;
  float* aproj = W + OFF_APROJ;
  float* qmean = W + OFF_QMEAN;
  float* qproj = W + OFF_QPROJ;
  float* x     = W + OFF_X;
  float* xn    = W + OFF_XN;
  float* xz    = W + OFF_XZ;
  float* xm    = W + OFF_XM;
  float* dtp   = W + OFF_DT;
  float* bpp   = W + OFF_BP;
  float* cpp   = W + OFF_CP;
  float* yss   = W + OFF_Y;
  float* yln   = W + OFF_YLN;
  float* y2    = W + OFF_Y2;
  float* rw    = W + OFF_RW;
  float* h     = xz;   // MoE hidden reuses xz (gate consumed before MoE)

  // ---- embeddings ----
  video_pool_k<<<(B_ * TV * CV * 32) / 256, 256, 0, stream>>>(video, vpool);
  qmean_k<<<(B_ * D + 255) / 256, 256, 0, stream>>>(q_emb, question, qmean);
  gemm_launch(vpool, vproj_w, vproj_b, vproj, nullptr, 0, B_ * TV, D, CV, 0, stream);
  gemm_launch(audio, aproj_w, aproj_b, aproj, nullptr, 0, B_ * TA, D, FA, 0, stream);
  gemm_launch(qmean, qproj_w, qproj_b, qproj, nullptr, 0, B_,      D, D,  0, stream);
  assemble_x_k<<<R, 256, 0, stream>>>(vproj, aproj, qproj, cls_tok, mod_emb, pos_emb, x);

  const int EW = R * INNER;
  for (int l = 0; l < NL; ++l) {
    // ---- mixer ----
    rmsnorm_k<<<R, 256, 0, stream>>>(x, norm1_w + (size_t)l * D, xn);
    gemm_launch(xn, in_w + (size_t)l * D * 2 * INNER, in_b + (size_t)l * 2 * INNER,
                xz, nullptr, 0, R, 2 * INNER, D, 0, stream);
    conv_silu_k<<<(EW + 255) / 256, 256, 0, stream>>>(
        xz, conv_w + (size_t)l * INNER * 3, conv_b + (size_t)l * INNER, xm);
    gemm_launch(xm, dt_w + (size_t)l * INNER * NS, dt_b + (size_t)l * NS, dtp, nullptr, 0, R, NS, INNER, 0, stream);
    gemm_launch(xm, Bp_w + (size_t)l * INNER * NS, Bp_b + (size_t)l * NS, bpp, nullptr, 0, R, NS, INNER, 0, stream);
    gemm_launch(xm, Cp_w + (size_t)l * INNER * NS, Cp_b + (size_t)l * NS, cpp, nullptr, 0, R, NS, INNER, 0, stream);
    ssm_scan_k<<<(B_ * NS + 255) / 256, 256, 0, stream>>>(dtp, bpp, cpp, yss);
    ln64_k<<<(R * 32 + 255) / 256, 256, 0, stream>>>(yss, yln);
    gemm_launch(yln, s2i_w + (size_t)l * NS * INNER, s2i_b + (size_t)l * INNER,
                y2, nullptr, 0, R, INNER, NS, 0, stream);
    gate_mix_k<<<(EW + 255) / 256, 256, 0, stream>>>(y2, xm, xz, D_par + (size_t)l * INNER);
    gemm_launch(y2, out_w + (size_t)l * INNER * D, out_b + (size_t)l * D,
                x, nullptr, 0, R, D, INNER, 1, stream);   // residual accumulate

    // ---- MoE (dense over experts, combine fused into expert-out epilogue) ----
    rmsnorm_k<<<R, 256, 0, stream>>>(x, norm2_w + (size_t)l * D, xn);
    routing_k<<<(R * 32 + 255) / 256, 256, 0, stream>>>(
        xn, gate_w + (size_t)l * D * E, gate_b + (size_t)l * E, rw);
    for (int e = 0; e < E; ++e) {
      const float* w1 = e_w1 + ((size_t)l * E + e) * D * HID;
      const float* b1 = e_b1 + ((size_t)l * E + e) * HID;
      const float* w2 = e_w2 + ((size_t)l * E + e) * HID * D;
      const float* b2 = e_b2 + ((size_t)l * E + e) * D;
      gemm_launch(xn, w1, b1, h, nullptr, 0, R, HID, D, 2, stream);       // gelu
      gemm_launch(h, w2, b2, x, rw + e, E, R, D, HID, 3, stream);         // x += w[:,e]*(h@w2+b2)
    }
  }

  head_k<<<B_, 256, 0, stream>>>(x, fnorm_w, head_w, head_b, (float*)d_out);
}
